// LightSelfAttention_27212912787904
// MI455X (gfx1250) — matrix operand order repaired and checked by reading, not compiled, not run on hardware
//
#include <hip/hip_runtime.h>
#include <hip/hip_bf16.h>

#define BB 8
#define SS 2048
#define DD 512
#define BSR (BB * SS)   // 16384 rows total

// B-tile staged in LDS: 64 rows (n) x 32 k-elements (bf16), row stride padded
// to 80 bytes (40 ushorts) so ds_load_b128 across 16 lanes is conflict-free.
#define ROWU 40

typedef __attribute__((ext_vector_type(16))) __bf16 v16bf;
typedef __attribute__((ext_vector_type(8)))  float  v8f;
typedef unsigned int   u32x4  __attribute__((ext_vector_type(4)));
typedef float          f32x4  __attribute__((ext_vector_type(4)));
typedef unsigned short u16x4  __attribute__((ext_vector_type(4)));
typedef int            i32x4  __attribute__((ext_vector_type(4)));

// address-space-qualified int4 for the async global->LDS DMA builtin
typedef i32x4 __attribute__((address_space(1))) gas_i32x4;
typedef i32x4 __attribute__((address_space(3))) las_i32x4;

union ABf { v16bf v; u32x4 u[2]; };

#if defined(__has_builtin)
#if __has_builtin(__builtin_amdgcn_global_load_async_to_lds_b128)
#define HAVE_ASYNC_COPY 1
#endif
#if __has_builtin(__builtin_amdgcn_s_wait_asynccnt)
#define HAVE_WAIT_ASYNC 1
#endif
#endif

__device__ __forceinline__ unsigned short f2bf(float f) {
    unsigned int u = __float_as_uint(f);
    unsigned int r = 0x7FFFu + ((u >> 16) & 1u);   // round-to-nearest-even
    return (unsigned short)((u + r) >> 16);
}
__device__ __forceinline__ float b2f(unsigned short h) {
    return __uint_as_float(((unsigned int)h) << 16);
}

__device__ __forceinline__ v8f wmma_bf16(const ABf& a, const ABf& b, v8f c) {
    return __builtin_amdgcn_wmma_f32_16x16x32_bf16(
        false, a.v, false, b.v, (short)0, c, false, false);
}

// 16-byte global -> LDS copy: async DMA when available, sync staging otherwise.
__device__ __forceinline__ void cp16(const unsigned short* g, unsigned short* l) {
#if defined(HAVE_ASYNC_COPY)
    __builtin_amdgcn_global_load_async_to_lds_b128(
        (gas_i32x4*)g, (las_i32x4*)l, 0, 0);
#else
    *(u32x4*)l = *(const u32x4*)g;
#endif
}

template <int N>
__device__ __forceinline__ void cp_wait() {
#if defined(HAVE_ASYNC_COPY)
#if defined(HAVE_WAIT_ASYNC)
    __builtin_amdgcn_s_wait_asynccnt(N);
#else
    asm volatile("s_wait_asynccnt %0" ::"i"(N) : "memory");
#endif
#endif
}

// ---------------------------------------------------------------------------
// fp32 -> bf16 convert, 4 elements per thread
// ---------------------------------------------------------------------------
__global__ void cvt_kernel(const float* __restrict__ in,
                           unsigned short* __restrict__ out, int n4) {
    int i = blockIdx.x * blockDim.x + threadIdx.x;
    if (i < n4) {
        f32x4 f = ((const f32x4*)in)[i];
        u16x4 o;
#pragma unroll
        for (int j = 0; j < 4; ++j) o[j] = f2bf(f[j]);
        ((u16x4*)out)[i] = o;
    }
}

// ---------------------------------------------------------------------------
// Projection: out[m,n] = sum_d x[m,d] * W[n,d] + bias[n]
// Block: M=256 (8 waves x 32 rows), N=64. B tile staged in LDS (double buf).
// transposed=1 stores out as [B, D, S] (for V).
// ---------------------------------------------------------------------------
__global__ void proj_kernel(const unsigned short* __restrict__ xb,
                            const unsigned short* __restrict__ wb,
                            const float* __restrict__ bias,
                            unsigned short* __restrict__ out,
                            int transposed) {
    __shared__ unsigned short btile[2][64 * ROWU];

    const int tid = threadIdx.x;
    const int lane = tid & 31, wave = tid >> 5;
    const int lo = lane & 15, hi = lane >> 4;
    const int mbase = blockIdx.x * 256 + wave * 32;
    const int nbase = blockIdx.y * 64;

    // cooperative B fill: thread -> (row fn, 8-ushort chunk fc)
    const int fn = tid >> 2;
    const int fc = tid & 3;
    const unsigned short* gbrow = wb + (size_t)(nbase + fn) * DD + fc * 8;

    v8f c[2][4];
#pragma unroll
    for (int h = 0; h < 2; ++h)
#pragma unroll
        for (int t = 0; t < 4; ++t)
#pragma unroll
            for (int v = 0; v < 8; ++v) c[h][t][v] = 0.0f;

    const unsigned short* arow0 = xb + (size_t)(mbase + lo) * DD;
    const unsigned short* arow1 = arow0 + (size_t)16 * DD;

    cp16(gbrow, &btile[0][fn * ROWU + fc * 8]);

    const int NK = DD / 32;
    for (int kc = 0; kc < NK; ++kc) {
        if (kc + 1 < NK) {
            cp16(gbrow + (kc + 1) * 32, &btile[(kc + 1) & 1][fn * ROWU + fc * 8]);
            cp_wait<1>();
        } else {
            cp_wait<0>();
        }
        __syncthreads();

        const int k = kc * 32;
        ABf a0, a1;
        a0.u[0] = *(const u32x4*)(arow0 + k + hi * 8);
        a0.u[1] = *(const u32x4*)(arow0 + k + 16 + hi * 8);
        a1.u[0] = *(const u32x4*)(arow1 + k + hi * 8);
        a1.u[1] = *(const u32x4*)(arow1 + k + 16 + hi * 8);
        const unsigned short* lb = btile[kc & 1];
#pragma unroll
        for (int t = 0; t < 4; ++t) {
            ABf b;
            const unsigned short* brow = lb + (t * 16 + lo) * ROWU + hi * 8;
            b.u[0] = *(const u32x4*)(brow);
            b.u[1] = *(const u32x4*)(brow + 16);
            c[0][t] = wmma_bf16(a0, b, c[0][t]);
            c[1][t] = wmma_bf16(a1, b, c[1][t]);
        }
        __syncthreads();
    }

#pragma unroll
    for (int h = 0; h < 2; ++h)
#pragma unroll
        for (int t = 0; t < 4; ++t) {
            const int n = nbase + t * 16 + lo;
            const float bv = bias[n];
#pragma unroll
            for (int v = 0; v < 8; ++v) {
                const int m = mbase + h * 16 + v + 8 * hi;
                const unsigned short val = f2bf(c[h][t][v] + bv);
                if (transposed) {
                    const int b_ = m >> 11, s_ = m & (SS - 1);
                    out[(size_t)(b_ * DD + n) * SS + s_] = val;
                } else {
                    out[(size_t)m * DD + n] = val;
                }
            }
        }
}

// ---------------------------------------------------------------------------
// scores[b,q,k] = (Q[b,q,:] . K[b,k,:]) / sqrt(D), key-masked, stored bf16
// ---------------------------------------------------------------------------
__global__ void scores_kernel(const unsigned short* __restrict__ qb,
                              const unsigned short* __restrict__ kb,
                              const int* __restrict__ mask,
                              unsigned short* __restrict__ sc) {
    __shared__ unsigned short btile[2][64 * ROWU];

    const int tid = threadIdx.x;
    const int lane = tid & 31, wave = tid >> 5;
    const int lo = lane & 15, hi = lane >> 4;
    const int gq = blockIdx.x * 256 + wave * 32;
    const int b = gq >> 11;
    const int q0 = gq & (SS - 1);
    const int nbase = blockIdx.y * 64;

    const int fn = tid >> 2;
    const int fc = tid & 3;
    const unsigned short* gbrow =
        kb + (size_t)(b * SS + nbase + fn) * DD + fc * 8;

    v8f c[2][4];
#pragma unroll
    for (int h = 0; h < 2; ++h)
#pragma unroll
        for (int t = 0; t < 4; ++t)
#pragma unroll
            for (int v = 0; v < 8; ++v) c[h][t][v] = 0.0f;

    const unsigned short* arow0 = qb + (size_t)(gq + lo) * DD;
    const unsigned short* arow1 = arow0 + (size_t)16 * DD;

    cp16(gbrow, &btile[0][fn * ROWU + fc * 8]);

    const int NK = DD / 32;
    for (int kc = 0; kc < NK; ++kc) {
        if (kc + 1 < NK) {
            cp16(gbrow + (kc + 1) * 32, &btile[(kc + 1) & 1][fn * ROWU + fc * 8]);
            cp_wait<1>();
        } else {
            cp_wait<0>();
        }
        __syncthreads();

        const int k = kc * 32;
        ABf a0, a1;
        a0.u[0] = *(const u32x4*)(arow0 + k + hi * 8);
        a0.u[1] = *(const u32x4*)(arow0 + k + 16 + hi * 8);
        a1.u[0] = *(const u32x4*)(arow1 + k + hi * 8);
        a1.u[1] = *(const u32x4*)(arow1 + k + 16 + hi * 8);
        const unsigned short* lb = btile[kc & 1];
#pragma unroll
        for (int t = 0; t < 4; ++t) {
            ABf bfr;
            const unsigned short* brow = lb + (t * 16 + lo) * ROWU + hi * 8;
            bfr.u[0] = *(const u32x4*)(brow);
            bfr.u[1] = *(const u32x4*)(brow + 16);
            c[0][t] = wmma_bf16(a0, bfr, c[0][t]);
            c[1][t] = wmma_bf16(a1, bfr, c[1][t]);
        }
        __syncthreads();
    }

    const float scale = 0.044194173824159220f;  // 1/sqrt(512)
#pragma unroll
    for (int h = 0; h < 2; ++h)
#pragma unroll
        for (int t = 0; t < 4; ++t) {
            const int key = nbase + t * 16 + lo;
            const bool on = mask[b * SS + key] != 0;
#pragma unroll
            for (int v = 0; v < 8; ++v) {
                const float val = on ? c[h][t][v] * scale : -10000.0f;
                sc[(size_t)(b * SS + q0 + h * 16 + v + 8 * hi) * SS + key] =
                    f2bf(val);
            }
        }
}

// ---------------------------------------------------------------------------
// In-place row softmax over 2048 bf16 values, fp32 math. One block per row.
// ---------------------------------------------------------------------------
__global__ void softmax_kernel(unsigned short* __restrict__ sc) {
    __shared__ float red[256];
    unsigned short* p = sc + (size_t)blockIdx.x * SS;
    const int tid = threadIdx.x;

    float r[8];
#pragma unroll
    for (int j = 0; j < 8; ++j) r[j] = b2f(p[tid + j * 256]);

    float m = r[0];
#pragma unroll
    for (int j = 1; j < 8; ++j) m = fmaxf(m, r[j]);
    red[tid] = m;
    __syncthreads();
    for (int s = 128; s > 0; s >>= 1) {
        if (tid < s) red[tid] = fmaxf(red[tid], red[tid + s]);
        __syncthreads();
    }
    m = red[0];
    __syncthreads();

    float sum = 0.0f;
#pragma unroll
    for (int j = 0; j < 8; ++j) {
        r[j] = __expf(r[j] - m);
        sum += r[j];
    }
    red[tid] = sum;
    __syncthreads();
    for (int s = 128; s > 0; s >>= 1) {
        if (tid < s) red[tid] += red[tid + s];
        __syncthreads();
    }
    const float inv = 1.0f / red[0];
#pragma unroll
    for (int j = 0; j < 8; ++j) p[tid + j * 256] = f2bf(r[j] * inv);
}

// ---------------------------------------------------------------------------
// out[b,q,n] = sum_k P[b,q,k] * V[b,k,n]   (V stored transposed vt[b,n,k])
// ---------------------------------------------------------------------------
__global__ void pv_kernel(const unsigned short* __restrict__ pb,
                          const unsigned short* __restrict__ vtb,
                          float* __restrict__ out) {
    __shared__ unsigned short btile[2][64 * ROWU];

    const int tid = threadIdx.x;
    const int lane = tid & 31, wave = tid >> 5;
    const int lo = lane & 15, hi = lane >> 4;
    const int gq = blockIdx.x * 256 + wave * 32;
    const int b = gq >> 11;
    const int q0 = gq & (SS - 1);
    const int nbase = blockIdx.y * 64;

    const int fn = tid >> 2;
    const int fc = tid & 3;
    const unsigned short* gbrow =
        vtb + (size_t)(b * DD + nbase + fn) * SS + fc * 8;

    v8f c[2][4];
#pragma unroll
    for (int h = 0; h < 2; ++h)
#pragma unroll
        for (int t = 0; t < 4; ++t)
#pragma unroll
            for (int v = 0; v < 8; ++v) c[h][t][v] = 0.0f;

    const unsigned short* arow0 = pb + (size_t)(b * SS + q0 + lo) * SS;
    const unsigned short* arow1 = arow0 + (size_t)16 * SS;

    cp16(gbrow, &btile[0][fn * ROWU + fc * 8]);

    const int NK = SS / 32;   // 64
    for (int kc = 0; kc < NK; ++kc) {
        if (kc + 1 < NK) {
            cp16(gbrow + (kc + 1) * 32, &btile[(kc + 1) & 1][fn * ROWU + fc * 8]);
            cp_wait<1>();
        } else {
            cp_wait<0>();
        }
        __syncthreads();

        const int k = kc * 32;
        ABf a0, a1;
        a0.u[0] = *(const u32x4*)(arow0 + k + hi * 8);
        a0.u[1] = *(const u32x4*)(arow0 + k + 16 + hi * 8);
        a1.u[0] = *(const u32x4*)(arow1 + k + hi * 8);
        a1.u[1] = *(const u32x4*)(arow1 + k + 16 + hi * 8);
        const unsigned short* lb = btile[kc & 1];
#pragma unroll
        for (int t = 0; t < 4; ++t) {
            ABf bfr;
            const unsigned short* brow = lb + (t * 16 + lo) * ROWU + hi * 8;
            bfr.u[0] = *(const u32x4*)(brow);
            bfr.u[1] = *(const u32x4*)(brow + 16);
            c[0][t] = wmma_bf16(a0, bfr, c[0][t]);
            c[1][t] = wmma_bf16(a1, bfr, c[1][t]);
        }
        __syncthreads();
    }

#pragma unroll
    for (int h = 0; h < 2; ++h)
#pragma unroll
        for (int t = 0; t < 4; ++t) {
            const int n = nbase + t * 16 + lo;
#pragma unroll
            for (int v = 0; v < 8; ++v) {
                out[(size_t)(gq + h * 16 + v + 8 * hi) * DD + n] = c[h][t][v];
            }
        }
}

// ---------------------------------------------------------------------------
extern "C" void kernel_launch(void* const* d_in, const int* in_sizes, int n_in,
                              void* d_out, int out_size, void* d_ws, size_t ws_size,
                              hipStream_t stream) {
    const float* x    = (const float*)d_in[0];
    const int*   mask = (const int*)d_in[1];
    const float* Wq   = (const float*)d_in[2];
    const float* bq   = (const float*)d_in[3];
    const float* Wk   = (const float*)d_in[4];
    const float* bk   = (const float*)d_in[5];
    const float* Wv   = (const float*)d_in[6];
    const float* bv   = (const float*)d_in[7];
    float* out = (float*)d_out;

    unsigned char* ws = (unsigned char*)d_ws;
    size_t off = 0;
    auto alloc = [&](size_t bytes) -> void* {
        void* p = ws + off;
        off += (bytes + 255) & ~(size_t)255;
        return p;
    };
    unsigned short* xb  = (unsigned short*)alloc((size_t)BSR * DD * 2);
    unsigned short* wqb = (unsigned short*)alloc((size_t)DD * DD * 2);
    unsigned short* wkb = (unsigned short*)alloc((size_t)DD * DD * 2);
    unsigned short* wvb = (unsigned short*)alloc((size_t)DD * DD * 2);
    unsigned short* qb  = (unsigned short*)alloc((size_t)BSR * DD * 2);
    unsigned short* kb  = (unsigned short*)alloc((size_t)BSR * DD * 2);
    unsigned short* vtb = (unsigned short*)alloc((size_t)BSR * DD * 2);
    unsigned short* sc  = (unsigned short*)alloc((size_t)BB * SS * SS * 2);
    (void)ws_size; (void)in_sizes; (void)n_in; (void)out_size;

    const int nx4 = (BSR * DD) / 4;
    cvt_kernel<<<(nx4 + 255) / 256, 256, 0, stream>>>(x, xb, nx4);
    const int nw4 = (DD * DD) / 4;
    cvt_kernel<<<(nw4 + 255) / 256, 256, 0, stream>>>(Wq, wqb, nw4);
    cvt_kernel<<<(nw4 + 255) / 256, 256, 0, stream>>>(Wk, wkb, nw4);
    cvt_kernel<<<(nw4 + 255) / 256, 256, 0, stream>>>(Wv, wvb, nw4);

    dim3 pg(BSR / 256, DD / 64);
    proj_kernel<<<pg, 256, 0, stream>>>(xb, wqb, bq, qb, 0);
    proj_kernel<<<pg, 256, 0, stream>>>(xb, wkb, bk, kb, 0);
    proj_kernel<<<pg, 256, 0, stream>>>(xb, wvb, bv, vtb, 1);

    dim3 sg(BSR / 256, SS / 64);
    scores_kernel<<<sg, 256, 0, stream>>>(qb, kb, mask, sc);

    softmax_kernel<<<BSR, 256, 0, stream>>>(sc);

    dim3 og(BSR / 256, DD / 64);
    pv_kernel<<<og, 256, 0, stream>>>(sc, vtb, out);
}
